// RAMSESModel_74560632259046
// MI455X (gfx1250) — 8-bit matrix operand order repaired and checked by reading, compiled, not run on hardware
//
#include <hip/hip_runtime.h>
#include <hip/hip_bf16.h>

typedef __attribute__((ext_vector_type(8))) int v8i;

#define NMASK 1536
#define HW    50176
#define KDIM  HW
#define POST_K 768
#define SIGMA_F 0.5f
#define THRESH_F 0.5f
#define EPS_F 1e-6f
#define LDSS  80
#define MAT_BYTES (128 * LDSS)
#define BUF_BYTES (2 * MAT_BYTES)
#define B_OFF MAT_BYTES

#if __has_builtin(__builtin_amdgcn_global_load_async_to_lds_b128)
#define HAVE_ASYNC_LDS 1
#else
#define HAVE_ASYNC_LDS 0
#endif

__global__ void k_convert_u8(const float* __restrict__ masks,
                             unsigned char* __restrict__ out, int n4) {
    int i = blockIdx.x * blockDim.x + threadIdx.x;
    if (i >= n4) return;
    float4 f = ((const float4*)masks)[i];
    uchar4 u;
    u.x = (unsigned char)f.x; u.y = (unsigned char)f.y;
    u.z = (unsigned char)f.z; u.w = (unsigned char)f.w;
    ((uchar4*)out)[i] = u;
}

__device__ __forceinline__ void async_copy16(const unsigned char* g, unsigned char* l) {
#if HAVE_ASYNC_LDS
    typedef int v4si __attribute__((__vector_size__(16)));
    typedef __attribute__((address_space(1))) v4si gv4si;
    typedef __attribute__((address_space(3))) v4si lv4si;
    __builtin_amdgcn_global_load_async_to_lds_b128(
        (gv4si*)(unsigned long long)(uintptr_t)g,
        (lv4si*)(unsigned int)(uintptr_t)l, 0, 0);
#else
    (void)g; (void)l;
#endif
}

__device__ __forceinline__ void wait_async0() {
#if HAVE_ASYNC_LDS
#if __has_builtin(__builtin_amdgcn_s_wait_asynccnt)
    __builtin_amdgcn_s_wait_asynccnt(0);
#else
    asm volatile("s_wait_asynccnt 0x0" ::: "memory");
#endif
#endif
}

union FragU8 {
    v8i  v;
    uint2 u2[4];
    uint4 u4[2];
};

__global__ __launch_bounds__(256)
void k_gemm_decay(const unsigned char* __restrict__ Au8,
                  const float* __restrict__ mask_sum,
                  const int*  __restrict__ labels,
                  float* __restrict__ decay) {
    __shared__ unsigned char lds[2 * BUF_BYTES];

    const int tj = blockIdx.x;
    const int ti = blockIdx.y;
    if (ti > tj) return;

    const int rowBase = ti * 128;
    const int colBase = tj * 128;
    const int tid  = threadIdx.x;
    const int lane = tid & 31;
    const int wid  = tid >> 5;
    const int waveM = wid & 3;
    const int waveN = wid >> 2;
    const int mrow = lane & 15;
    const int half = lane >> 4;

    v8i acc[2][4];
    const v8i vzero = {0,0,0,0,0,0,0,0};
#pragma unroll
    for (int a = 0; a < 2; ++a)
#pragma unroll
        for (int b = 0; b < 4; ++b) acc[a][b] = vzero;

    const int fr = (tid * 16) >> 6;
    const int fc = (tid * 16) & 63;
    const unsigned char* gA0 = Au8 + (size_t)(rowBase + fr) * KDIM + fc;
    const unsigned char* gA1 = Au8 + (size_t)(rowBase + fr + 64) * KDIM + fc;
    const unsigned char* gB0 = Au8 + (size_t)(colBase + fr) * KDIM + fc;
    const unsigned char* gB1 = Au8 + (size_t)(colBase + fr + 64) * KDIM + fc;
    const int la0 = fr * LDSS + fc;
    const int la1 = (fr + 64) * LDSS + fc;

    const int aRow0 = (waveM * 32 + mrow) * LDSS + half * 16;
    const int bRow0 = B_OFF + (waveN * 64 + mrow) * LDSS + half * 16;

    const int NSTEP = KDIM / 64;

    unsigned curOff = 0;

#if HAVE_ASYNC_LDS
    async_copy16(gA0, &lds[la0]);
    async_copy16(gA1, &lds[la1]);
    async_copy16(gB0, &lds[B_OFF + la0]);
    async_copy16(gB1, &lds[B_OFF + la1]);
    wait_async0();
    __syncthreads();
#else
    {
        uint4 ra0 = *(const uint4*)gA0, ra1 = *(const uint4*)gA1;
        uint4 rb0 = *(const uint4*)gB0, rb1 = *(const uint4*)gB1;
        *(uint4*)&lds[la0] = ra0;         *(uint4*)&lds[la1] = ra1;
        *(uint4*)&lds[B_OFF + la0] = rb0; *(uint4*)&lds[B_OFF + la1] = rb1;
    }
    __syncthreads();
#endif

#pragma clang loop unroll(disable)
    for (int step = 0; step < NSTEP; ++step) {
        const unsigned nxtOff = curOff ^ BUF_BYTES;
        const int kbn = (step + 1) * 64;
        const bool more = (step + 1 < NSTEP);

#if HAVE_ASYNC_LDS
        if (more) {
            async_copy16(gA0 + kbn, &lds[nxtOff + la0]);
            async_copy16(gA1 + kbn, &lds[nxtOff + la1]);
            async_copy16(gB0 + kbn, &lds[nxtOff + B_OFF + la0]);
            async_copy16(gB1 + kbn, &lds[nxtOff + B_OFF + la1]);
        }
#else
        uint4 ra0, ra1, rb0, rb1;
        if (more) {
            ra0 = *(const uint4*)(gA0 + kbn); ra1 = *(const uint4*)(gA1 + kbn);
            rb0 = *(const uint4*)(gB0 + kbn); rb1 = *(const uint4*)(gB1 + kbn);
        }
#endif

        FragU8 aF[2];
#pragma unroll
        for (int mt = 0; mt < 2; ++mt) {
            const unsigned char* base = &lds[curOff + aRow0 + mt * 16 * LDSS];
            aF[mt].u4[0] = *(const uint4*)(base + 0);
            aF[mt].u4[1] = *(const uint4*)(base + 32);
        }
        FragU8 bF[4];
#pragma unroll
        for (int nt = 0; nt < 4; ++nt) {
            const unsigned char* base = &lds[curOff + bRow0 + nt * 16 * LDSS];
            bF[nt].u4[0] = *(const uint4*)(base + 0);
            bF[nt].u4[1] = *(const uint4*)(base + 32);
        }
#pragma unroll
        for (int mt = 0; mt < 2; ++mt)
#pragma unroll
            for (int nt = 0; nt < 4; ++nt)
                acc[mt][nt] = __builtin_amdgcn_wmma_i32_16x16x64_iu8(
                    false, aF[mt].v, false, bF[nt].v, acc[mt][nt], false, false);

#if HAVE_ASYNC_LDS
        wait_async0();
        __syncthreads();
#else
        if (more) {
            *(uint4*)&lds[nxtOff + la0] = ra0;         *(uint4*)&lds[nxtOff + la1] = ra1;
            *(uint4*)&lds[nxtOff + B_OFF + la0] = rb0; *(uint4*)&lds[nxtOff + B_OFF + la1] = rb1;
        }
        __syncthreads();
#endif
        curOff = nxtOff;
    }

#pragma unroll
    for (int mt = 0; mt < 2; ++mt) {
#pragma unroll
        for (int nt = 0; nt < 4; ++nt) {
#pragma unroll
            for (int r = 0; r < 8; ++r) {
                int i = rowBase + waveM * 32 + mt * 16 + r + 8 * half;
                int j = colBase + waveN * 64 + nt * 16 + mrow;
                float inter = (float)acc[mt][nt][r];
                float uni = 2.0f * mask_sum[j] - inter;
                float iou = inter / (uni + EPS_F);
                float d = (j > i && labels[i] == labels[j]) ? iou : 0.0f;
                decay[(size_t)i * NMASK + j] = d;
            }
        }
    }
}

__global__ void k_colmax(const float* __restrict__ decay, float* __restrict__ comp) {
    int j = blockIdx.x * blockDim.x + threadIdx.x;
    if (j >= NMASK) return;
    float m = -3.4e38f;
    for (int i = 0; i < NMASK; ++i)
        m = fmaxf(m, decay[(size_t)i * NMASK + j]);
    comp[j] = m;
}

__global__ void k_decaycoef(const float* __restrict__ decay,
                            const float* __restrict__ comp,
                            const float* __restrict__ scores,
                            float* __restrict__ key,
                            float* __restrict__ validf) {
    int j = blockIdx.x * blockDim.x + threadIdx.x;
    if (j >= NMASK) return;
    float t = 3.4e38f;
    for (int i = 0; i < NMASK; ++i) {
        float c = comp[i];
        float d = decay[(size_t)i * NMASK + j];
        t = fminf(t, c * c - d * d);
    }
    float coeff = expf(t / SIGMA_F);
    float dscore = scores[j] * coeff;
    int valid = dscore >= THRESH_F;
    validf[j] = valid ? 1.0f : 0.0f;
    key[j] = valid ? scores[j] : -1.0f;
}

__global__ void k_select(const float* __restrict__ key,
                         const float* __restrict__ scores,
                         const int*  __restrict__ labels,
                         const float* __restrict__ factors,
                         const float* __restrict__ validf,
                         int*  __restrict__ sel_idx,
                         float* __restrict__ sel_valid,
                         float* __restrict__ out_scores,
                         float* __restrict__ out_labels,
                         float* __restrict__ out_factors) {
    __shared__ float sk[NMASK];
    int tid = threadIdx.x;
    for (int i = tid; i < NMASK; i += blockDim.x) sk[i] = key[i];
    __syncthreads();
    int j = blockIdx.x * blockDim.x + tid;
    if (j >= NMASK) return;
    float kj = sk[j];
    int rank = 0;
    for (int i = 0; i < NMASK; ++i) {
        float ki = sk[i];
        rank += (ki > kj) || (ki == kj && i < j);
    }
    if (rank < POST_K) {
        bool v = validf[j] > 0.5f;
        sel_idx[rank]   = j;
        sel_valid[rank] = v ? 1.0f : 0.0f;
        out_scores[rank]  = v ? scores[j] : 0.0f;
        out_labels[rank]  = v ? (float)labels[j] : -1.0f;
        out_factors[rank] = v ? factors[j] : 0.0f;
    }
}

__global__ void k_gather_seg(const float* __restrict__ seg,
                             const int*  __restrict__ sel_idx,
                             const float* __restrict__ sel_valid,
                             float* __restrict__ out_seg) {
    int c4 = blockIdx.x * blockDim.x + threadIdx.x;
    int r  = blockIdx.y;
    if (c4 >= HW / 4) return;
    int idx = sel_idx[r];
    bool kept = sel_valid[r] > 0.5f;
    float4 z = make_float4(0.f, 0.f, 0.f, 0.f);
    float4 val = kept ? ((const float4*)(seg + (size_t)idx * HW))[c4] : z;
    ((float4*)(out_seg + (size_t)r * HW))[c4] = val;
}

extern "C" void kernel_launch(void* const* d_in, const int* in_sizes, int n_in,
                              void* d_out, int out_size, void* d_ws, size_t ws_size,
                              hipStream_t stream) {
    const int*   cls_labels   = (const int*)  d_in[0];
    const float* scores       = (const float*)d_in[1];
    const float* cls_factors  = (const float*)d_in[2];
    const float* seg_preds    = (const float*)d_in[3];
    const float* binary_masks = (const float*)d_in[4];
    const float* mask_sum     = (const float*)d_in[5];

    unsigned char* ws = (unsigned char*)d_ws;
    size_t o = 0;
    unsigned char* mask_u8 = ws + o; o += (size_t)NMASK * KDIM;
    float* decay  = (float*)(ws + o); o += (size_t)NMASK * NMASK * 4;
    float* comp   = (float*)(ws + o); o += NMASK * 4;
    float* key    = (float*)(ws + o); o += NMASK * 4;
    float* validf = (float*)(ws + o); o += NMASK * 4;
    int*   sel_idx   = (int*)(ws + o); o += POST_K * 4;
    float* sel_valid = (float*)(ws + o); o += POST_K * 4;

    float* out = (float*)d_out;
    float* out_seg     = out;
    float* out_scores  = out + (size_t)POST_K * HW;
    float* out_labels  = out_scores + POST_K;
    float* out_factors = out_labels + POST_K;

    int n4 = NMASK * KDIM / 4;
    k_convert_u8<<<(n4 + 255) / 256, 256, 0, stream>>>(binary_masks, mask_u8, n4);

    (void)hipMemsetAsync(decay, 0, (size_t)NMASK * NMASK * 4, stream);

    dim3 g(NMASK / 128, NMASK / 128);
    k_gemm_decay<<<g, 256, 0, stream>>>(mask_u8, mask_sum, cls_labels, decay);

    k_colmax<<<NMASK / 256, 256, 0, stream>>>(decay, comp);

    k_decaycoef<<<NMASK / 256, 256, 0, stream>>>(decay, comp, scores, key, validf);

    k_select<<<NMASK / 256, 256, 0, stream>>>(key, scores, cls_labels, cls_factors,
                                              validf, sel_idx, sel_valid,
                                              out_scores, out_labels, out_factors);

    dim3 gg(HW / 4 / 256, POST_K);
    k_gather_seg<<<gg, 256, 0, stream>>>(seg_preds, sel_idx, sel_valid, out_seg);
}
